// GraphTransformer_10591389352275
// MI455X (gfx1250) — matrix operand order repaired and checked by reading, not compiled, not run on hardware
//
#include <hip/hip_runtime.h>

#define NN      50000
#define EE      600000
#define FIN     64
#define HID     128
#define HEADS   8
#define DH      16
#define NCLS    10
#define NLAYERS 3

typedef __attribute__((ext_vector_type(16))) _Float16 v16h;
typedef __attribute__((ext_vector_type(8)))  _Float16 v8h;
typedef __attribute__((ext_vector_type(8)))  float    v8f;

// ---------------------------------------------------------------------------
// fp32 -> f16 convert
__global__ void k_cvt_f16(const float* __restrict__ s, _Float16* __restrict__ d, int n) {
  int i = blockIdx.x * blockDim.x + threadIdx.x;
  if (i < n) d[i] = (_Float16)s[i];
}

// ---------------------------------------------------------------------------
// Pack a row-major fp32 weight W[K x 128] into per-lane WMMA B fragments (f16).
// Fragment layout (16-bit B 32x16, contiguous K order, same as A): lane holds col = 16*ntile + (lane&15),
// elements j = W[ks*32 + 16*(lane>>4) + j][col], j = 0..15.
__global__ void k_pack_b(const float* __restrict__ W, _Float16* __restrict__ Bp, int K) {
  int idx = blockIdx.x * blockDim.x + threadIdx.x;
  int ksteps = K >> 5;
  int total = 8 * ksteps * 32;
  if (idx >= total) return;
  int lane  = idx & 31;
  int ks    = (idx >> 5) % ksteps;
  int ntile = (idx >> 5) / ksteps;
  int col = ntile * 16 + (lane & 15);
  int kb  = ks * 32 + ((lane >> 4) * 16);
  _Float16* o = Bp + (size_t)idx * 16;
#pragma unroll
  for (int j = 0; j < 16; ++j)
    o[j] = (_Float16)W[(size_t)(kb + j) * HID + col];
}

// ---------------------------------------------------------------------------
// WMMA GEMM: out[M x 128] = A[M x K](f16,row-major) * Bpack + bias, optional relu.
// Block = 256 threads = 8 waves; block tile 64 rows x 128 cols.
// Wave w: rows base + 16*(w>>1), cols 64*(w&1); 4 accumulators of 16x16.
// K templated (64/128) so the k-loop fully unrolls: all A fragments preloaded,
// per-kstep the 4 B fragments are clause-loaded before the 4 WMMAs.
template <int K>
__global__ __launch_bounds__(256) void k_gemm_wmma(
    const _Float16* __restrict__ A, const _Float16* __restrict__ Bp,
    const float* __restrict__ bias, float* __restrict__ outF,
    _Float16* __restrict__ outH, int M, int doRelu)
{
  constexpr int KSTEPS = K >> 5;
  const int lane    = threadIdx.x & 31;
  const int wave    = threadIdx.x >> 5;
  const int rowBase = blockIdx.x * 64 + (wave >> 1) * 16;
  const int colBase = (wave & 1) * 64;
  if (rowBase >= M) return;                 // wave-uniform: EXEC stays all-ones

  v8f acc[4];
#pragma unroll
  for (int t = 0; t < 4; ++t) acc[t] = (v8f){0.f,0.f,0.f,0.f,0.f,0.f,0.f,0.f};

  int rowA = rowBase + (lane & 15);
  if (rowA >= M) rowA = M - 1;
  const int kSel = (lane >> 4) * 16;        // lane half selects K sub-range

  // Preload all A fragments (contiguous K order: lanes 0-15 hold K 0-15, 16-31 hold 16-31)
  v16h av[KSTEPS];
#pragma unroll
  for (int ks = 0; ks < KSTEPS; ++ks) {
    const _Float16* arow = A + (size_t)rowA * K + ks * 32 + kSel;
    v8h alo = *reinterpret_cast<const v8h*>(arow);
    v8h ahi = *reinterpret_cast<const v8h*>(arow + 8);
#pragma unroll
    for (int j = 0; j < 8; ++j) { av[ks][j] = alo[j]; av[ks][j + 8] = ahi[j]; }
  }

#pragma unroll
  for (int ks = 0; ks < KSTEPS; ++ks) {
    v16h bf[4];
#pragma unroll
    for (int t = 0; t < 4; ++t) {
      const _Float16* bp_ =
          Bp + (((size_t)((colBase >> 4) + t) * KSTEPS + ks) * 32 + lane) * 16;
      v8h blo = *reinterpret_cast<const v8h*>(bp_);
      v8h bhi = *reinterpret_cast<const v8h*>(bp_ + 8);
#pragma unroll
      for (int j = 0; j < 8; ++j) { bf[t][j] = blo[j]; bf[t][j + 8] = bhi[j]; }
    }
#pragma unroll
    for (int t = 0; t < 4; ++t)
      acc[t] = __builtin_amdgcn_wmma_f32_16x16x32_f16(
          false, av[ks], false, bf[t], (short)0, acc[t], false, false);
  }

  // D layout: element r <-> row rowBase + 8*(lane>>4) + r, col = colBase+16*t+(lane&15)
  int rbase = rowBase + (lane >> 4) * 8;
  int cl = lane & 15;
#pragma unroll
  for (int t = 0; t < 4; ++t) {
    int col = colBase + t * 16 + cl;
    float bb = bias ? bias[col] : 0.f;
#pragma unroll
    for (int r = 0; r < 8; ++r) {
      int row = rbase + r;
      if (row < M) {
        float val = acc[t][r] + bb;
        if (doRelu) val = fmaxf(val, 0.f);
        if (outF) outF[(size_t)row * HID + col] = val;
        if (outH) outH[(size_t)row * HID + col] = (_Float16)val;
      }
    }
  }
}

// ---------------------------------------------------------------------------
// CSR-by-destination build (once per launch; graph is layer-invariant).
__global__ void k_zero_u32(unsigned* __restrict__ a, unsigned* __restrict__ b, int n) {
  int i = blockIdx.x * blockDim.x + threadIdx.x;
  if (i < n) { a[i] = 0u; b[i] = 0u; }
}

__global__ void k_count(const int* __restrict__ dst, unsigned* __restrict__ cnt) {
  int e = blockIdx.x * blockDim.x + threadIdx.x;
  if (e < EE) atomicAdd(&cnt[dst[e]], 1u);
}

// Exclusive scan over NN counts -> off[NN+1]; single 1024-thread block.
__global__ __launch_bounds__(1024) void k_scan_offsets(const unsigned* __restrict__ cnt,
                                                       unsigned* __restrict__ off) {
  __shared__ unsigned tmp[1024];
  __shared__ unsigned carry;
  if (threadIdx.x == 0) carry = 0u;
  __syncthreads();
  for (int base = 0; base < NN; base += 1024) {
    int i = base + (int)threadIdx.x;
    unsigned val = (i < NN) ? cnt[i] : 0u;
    tmp[threadIdx.x] = val;
    __syncthreads();
    for (int d = 1; d < 1024; d <<= 1) {
      unsigned t = (threadIdx.x >= (unsigned)d) ? tmp[threadIdx.x - d] : 0u;
      __syncthreads();
      tmp[threadIdx.x] += t;
      __syncthreads();
    }
    unsigned incl = tmp[threadIdx.x];
    if (i < NN) off[i] = carry + (incl - val);
    __syncthreads();
    if (threadIdx.x == 1023) carry += tmp[1023];
    __syncthreads();
  }
  if (threadIdx.x == 0) off[NN] = carry;
}

__global__ void k_fill_csr(const int* __restrict__ src, const int* __restrict__ dst,
                           const float* __restrict__ dist, const unsigned* __restrict__ off,
                           unsigned* __restrict__ cursor, int* __restrict__ eid,
                           int* __restrict__ csrc, float* __restrict__ cdist) {
  int e = blockIdx.x * blockDim.x + threadIdx.x;
  if (e >= EE) return;
  int d = dst[e];
  unsigned pos = off[d] + atomicAdd(&cursor[d], 1u);
  eid[pos]   = e;
  csrc[pos]  = src[e];
  cdist[pos] = dist[e];
}

// Canonicalize bucket order by edge id (insertion sort, avg degree 12) so the
// floating-point reduction order -- and thus the output -- is bitwise
// deterministic across replays (float atomics would not be).
__global__ void k_sort_csr(const unsigned* __restrict__ off, int* __restrict__ eid,
                           int* __restrict__ csrc, float* __restrict__ cdist) {
  int n = blockIdx.x * blockDim.x + threadIdx.x;
  if (n >= NN) return;
  int lo = (int)off[n], hi = (int)off[n + 1];
  for (int i = lo + 1; i < hi; ++i) {
    int ke = eid[i], ks = csrc[i];
    float kd = cdist[i];
    int j = i - 1;
    while (j >= lo && eid[j] > ke) {
      eid[j + 1] = eid[j]; csrc[j + 1] = csrc[j]; cdist[j + 1] = cdist[j];
      --j;
    }
    eid[j + 1] = ke; csrc[j + 1] = ks; cdist[j + 1] = kd;
  }
}

// ---------------------------------------------------------------------------
// Fused per-node attention aggregation (flash-attention-style online softmax).
// One wave32 per node; lane l owns dims [4l, 4l+4); head h = lanes 4h..4h+3.
// Per edge: alpha = q . (k[src] + dist*We) / 4 via 2 shfl_xor reductions within
// the 4-lane head group; running max/sum/accumulator rescaled in registers.
// Zero atomics; k/v gathers are 512B coalesced rows (L2-resident).
__global__ __launch_bounds__(256) void k_attn(
    const float* __restrict__ q, const float* __restrict__ kM,
    const float* __restrict__ vM, const float* __restrict__ We,
    const unsigned* __restrict__ off, const int* __restrict__ csrc,
    const float* __restrict__ cdist, float* __restrict__ out /* holds skip */) {
  int lane = threadIdx.x & 31;
  int node = blockIdx.x * 8 + (threadIdx.x >> 5);
  if (node >= NN) return;                      // wave-uniform

  float4 q4 = *(const float4*)(q + (size_t)node * HID + lane * 4);
  float4 w4 = *(const float4*)(We + lane * 4);

  float m = -3.0e38f, s = 0.f;
  float4 acc = {0.f, 0.f, 0.f, 0.f};
  int lo = (int)off[node], hi = (int)off[node + 1];

  for (int j = lo; j < hi; ++j) {
    int sIdx = csrc[j];
    float dd = cdist[j];
    if (j + 1 < hi) {
      int sn = csrc[j + 1];
      __builtin_prefetch(kM + (size_t)sn * HID + lane * 4, 0, 3);
      __builtin_prefetch(vM + (size_t)sn * HID + lane * 4, 0, 3);
    }
    float4 k4 = *(const float4*)(kM + (size_t)sIdx * HID + lane * 4);
    float kex = k4.x + dd * w4.x, key = k4.y + dd * w4.y;
    float kez = k4.z + dd * w4.z, kew = k4.w + dd * w4.w;
    float part = q4.x * kex + q4.y * key + q4.z * kez + q4.w * kew;
    part += __shfl_xor(part, 1, 32);           // reduce within 4-lane head group
    part += __shfl_xor(part, 2, 32);
    float alpha = part * 0.25f;                // 1/sqrt(DH)

    float mn    = fmaxf(m, alpha);
    float scale = __expf(m - mn);
    float p     = __expf(alpha - mn);
    s = s * scale + p;
    float4 v4 = *(const float4*)(vM + (size_t)sIdx * HID + lane * 4);
    acc.x = acc.x * scale + p * (v4.x + dd * w4.x);
    acc.y = acc.y * scale + p * (v4.y + dd * w4.y);
    acc.z = acc.z * scale + p * (v4.z + dd * w4.z);
    acc.w = acc.w * scale + p * (v4.w + dd * w4.w);
    m = mn;
  }

  float inv = 1.f / (s + 1e-16f);
  float* op = out + (size_t)node * HID + lane * 4;
  float4 o4 = *(const float4*)op;              // skip connection (h @ Ws + bs)
  o4.x += acc.x * inv; o4.y += acc.y * inv;
  o4.z += acc.z * inv; o4.w += acc.w * inv;
  *(float4*)op = o4;
}

// ---------------------------------------------------------------------------
// h16 = f16(relu(layernorm(in)))   (one node per thread; rows are L2 hot)
__global__ void k_ln_relu(const float* __restrict__ in, const float* __restrict__ g,
                          const float* __restrict__ b, _Float16* __restrict__ h16) {
  int n = blockIdx.x * blockDim.x + threadIdx.x;
  if (n >= NN) return;
  const float* r = in + (size_t)n * HID;
  float s = 0.f, s2 = 0.f;
#pragma unroll 8
  for (int j = 0; j < HID; ++j) { float xv = r[j]; s += xv; s2 += xv * xv; }
  float m = s * (1.f / HID);
  float var = s2 * (1.f / HID) - m * m;
  float inv = rsqrtf(var + 1e-5f);
#pragma unroll 8
  for (int j = 0; j < HID; ++j) {
    float y = (r[j] - m) * inv * g[j] + b[j];
    h16[(size_t)n * HID + j] = (_Float16)fmaxf(y, 0.f);
  }
}

// final: relu(LN(in)) @ W_after + b_after, log_softmax -> out[N x 10]
__global__ void k_final(const float* __restrict__ in, const float* __restrict__ g,
                        const float* __restrict__ b, const float* __restrict__ Wa,
                        const float* __restrict__ ba, float* __restrict__ out) {
  int n = blockIdx.x * blockDim.x + threadIdx.x;
  if (n >= NN) return;
  const float* r = in + (size_t)n * HID;
  float s = 0.f, s2 = 0.f;
#pragma unroll 8
  for (int j = 0; j < HID; ++j) { float xv = r[j]; s += xv; s2 += xv * xv; }
  float m = s * (1.f / HID);
  float inv = rsqrtf(s2 * (1.f / HID) - m * m + 1e-5f);
  float logit[NCLS];
#pragma unroll
  for (int c = 0; c < NCLS; ++c) logit[c] = ba[c];
  for (int j = 0; j < HID; ++j) {
    float y = fmaxf((r[j] - m) * inv * g[j] + b[j], 0.f);
#pragma unroll
    for (int c = 0; c < NCLS; ++c) logit[c] += y * Wa[j * NCLS + c];
  }
  float mx = logit[0];
#pragma unroll
  for (int c = 1; c < NCLS; ++c) mx = fmaxf(mx, logit[c]);
  float sum = 0.f;
#pragma unroll
  for (int c = 0; c < NCLS; ++c) sum += __expf(logit[c] - mx);
  float lse = mx + __logf(sum);
#pragma unroll
  for (int c = 0; c < NCLS; ++c) out[(size_t)n * NCLS + c] = logit[c] - lse;
}

// ---------------------------------------------------------------------------
extern "C" void kernel_launch(void* const* d_in, const int* in_sizes, int n_in,
                              void* d_out, int out_size, void* d_ws, size_t ws_size,
                              hipStream_t stream) {
  (void)in_sizes; (void)n_in; (void)out_size; (void)ws_size;
  const float* x        = (const float*)d_in[0];
  const int*   edge     = (const int*)  d_in[1];
  const float* dist     = (const float*)d_in[2];
  const float* Wq       = (const float*)d_in[3];
  const float* bq       = (const float*)d_in[4];
  const float* Wk       = (const float*)d_in[5];
  const float* bk       = (const float*)d_in[6];
  const float* Wv       = (const float*)d_in[7];
  const float* bv       = (const float*)d_in[8];
  const float* WeA      = (const float*)d_in[9];
  const float* Ws       = (const float*)d_in[10];
  const float* bs       = (const float*)d_in[11];
  const float* ln_g     = (const float*)d_in[12];
  const float* ln_b     = (const float*)d_in[13];
  const float* W_before = (const float*)d_in[14];
  const float* b_before = (const float*)d_in[15];
  const float* W_after  = (const float*)d_in[16];
  const float* b_after  = (const float*)d_in[17];
  const int* srcI = edge;
  const int* dstI = edge + EE;

  char* ws = (char*)d_ws;
  size_t off_ = 0;
  auto take = [&](size_t bytes) -> char* {
    char* p = ws + off_;
    off_ = (off_ + bytes + 255) & ~(size_t)255;
    return p;
  };
  _Float16* h16    = (_Float16*)take((size_t)NN * HID * 2);
  _Float16* x16    = (_Float16*)take((size_t)NN * FIN * 2);
  float*    qB     = (float*)take((size_t)NN * HID * 4);
  float*    kB     = (float*)take((size_t)NN * HID * 4);
  float*    vB     = (float*)take((size_t)NN * HID * 4);
  float*    outB   = (float*)take((size_t)NN * HID * 4);
  unsigned* cnt    = (unsigned*)take((size_t)NN * 4);
  unsigned* cursor = (unsigned*)take((size_t)NN * 4);
  unsigned* offs   = (unsigned*)take((size_t)(NN + 1) * 4);
  int*      eid    = (int*)take((size_t)EE * 4);
  int*      csrc   = (int*)take((size_t)EE * 4);
  float*    cdist  = (float*)take((size_t)EE * 4);
  _Float16* packWb = (_Float16*)take((size_t)FIN * HID * 2);
  _Float16* packL  = (_Float16*)take((size_t)NLAYERS * 4 * HID * HID * 2);

  const int TPB = 256;
  const int nBlocks  = (NN + TPB - 1) / TPB;
  const int eBlocks  = (EE + TPB - 1) / TPB;

  // ---- once per launch: convert x, pack weights, build deterministic CSR --
  { int tot = NN * FIN;
    k_cvt_f16<<<(tot + TPB - 1) / TPB, TPB, 0, stream>>>(x, x16, tot); }
  { int tot = 8 * (FIN / 32) * 32;
    k_pack_b<<<(tot + TPB - 1) / TPB, TPB, 0, stream>>>(W_before, packWb, FIN); }
  for (int l = 0; l < NLAYERS; ++l) {
    const float* Wm[4] = { Wq + (size_t)l * HID * HID, Wk + (size_t)l * HID * HID,
                           Wv + (size_t)l * HID * HID, Ws + (size_t)l * HID * HID };
    for (int w = 0; w < 4; ++w) {
      int tot = 8 * (HID / 32) * 32;
      k_pack_b<<<(tot + TPB - 1) / TPB, TPB, 0, stream>>>(
          Wm[w], packL + ((size_t)(l * 4 + w)) * HID * HID, HID);
    }
  }
  k_zero_u32<<<nBlocks, TPB, 0, stream>>>(cnt, cursor, NN);
  k_count<<<eBlocks, TPB, 0, stream>>>(dstI, cnt);
  k_scan_offsets<<<1, 1024, 0, stream>>>(cnt, offs);
  k_fill_csr<<<eBlocks, TPB, 0, stream>>>(srcI, dstI, dist, offs, cursor, eid, csrc, cdist);
  k_sort_csr<<<nBlocks, TPB, 0, stream>>>(offs, eid, csrc, cdist);

  const int gemmBlocks = (NN + 63) / 64;
  // ---- fcnn_before: h = relu(x @ W_before + b) -> f16 ------------------
  k_gemm_wmma<FIN><<<gemmBlocks, TPB, 0, stream>>>(x16, packWb, b_before,
                                                   nullptr, h16, NN, 1);

  const int attnBlocks = (NN + 7) / 8;
  for (int l = 0; l < NLAYERS; ++l) {
    _Float16* pq = packL + ((size_t)(l * 4 + 0)) * HID * HID;
    _Float16* pk = packL + ((size_t)(l * 4 + 1)) * HID * HID;
    _Float16* pv = packL + ((size_t)(l * 4 + 2)) * HID * HID;
    _Float16* ps = packL + ((size_t)(l * 4 + 3)) * HID * HID;
    k_gemm_wmma<HID><<<gemmBlocks, TPB, 0, stream>>>(h16, pq, bq + l * HID, qB,   nullptr, NN, 0);
    k_gemm_wmma<HID><<<gemmBlocks, TPB, 0, stream>>>(h16, pk, bk + l * HID, kB,   nullptr, NN, 0);
    k_gemm_wmma<HID><<<gemmBlocks, TPB, 0, stream>>>(h16, pv, bv + l * HID, vB,   nullptr, NN, 0);
    k_gemm_wmma<HID><<<gemmBlocks, TPB, 0, stream>>>(h16, ps, bs + l * HID, outB, nullptr, NN, 0);

    k_attn<<<attnBlocks, TPB, 0, stream>>>(qB, kB, vB, WeA + l * HID,
                                           offs, csrc, cdist, outB);
    if (l < NLAYERS - 1)
      k_ln_relu<<<nBlocks, TPB, 0, stream>>>(outB, ln_g + l * HID, ln_b + l * HID, h16);
  }

  k_final<<<nBlocks, TPB, 0, stream>>>(outB, ln_g + 2 * HID, ln_b + 2 * HID,
                                       W_after, b_after, (float*)d_out);
}
